// AttentionMap_24661702214349
// MI455X (gfx1250) — compile-verified, matrix operand order checked by reading, not run on hardware
//
#include <hip/hip_runtime.h>
#include <hip/hip_bf16.h>

// ---------------------------------------------------------------------------
// AttentionMap for MI455X (gfx1250), bf16 WMMA + async-to-LDS bias staging.
//   cvt   : one-shot f32->bf16 of x, W_up, W_proj into ws (L2-resident)
//   gemm1 : xe[1024,2560] = x @ W_up^T      (16x64 tile/wave, 4 WMMA/K-step)
//   gemm2 : t[1024,5120]  = xe @ W_proj^T, scattered into q_ws/k_ws
//           [B,H,L,32] bf16 (c padded 20->32 with zeros, q pre-scaled)
//   attn  : per (b,h): one 16x16x32 WMMA per q/k tile, bias tile DMA'd
//           global->LDS asynchronously under the WMMA, fused mask + store.
// ---------------------------------------------------------------------------

typedef __bf16 v16bf __attribute__((ext_vector_type(16)));
typedef float  v8f   __attribute__((ext_vector_type(8)));
typedef int    v4i   __attribute__((ext_vector_type(4)));

#define AS1 __attribute__((address_space(1)))
#define AS3 __attribute__((address_space(3)))

constexpr int B  = 2;
constexpr int L  = 512;
constexpr int T  = 1024;
constexpr int E  = 2560;
constexpr int H  = 128;
constexpr int HW = 20;        // E / H
constexpr int F  = 2 * E;     // 5120
constexpr int M  = B * L;     // 1024
constexpr int CP = 32;        // padded head dim (WMMA K)

constexpr size_t X16_BYTES  = (size_t)M * T * 2;           //  2.0 MB
constexpr size_t WU16_BYTES = (size_t)E * T * 2;           //  5.2 MB
constexpr size_t WP16_BYTES = (size_t)F * E * 2;           // 26.2 MB
constexpr size_t XE_BYTES   = (size_t)M * E * 2;           //  5.2 MB
constexpr size_t QK_BYTES   = (size_t)B * H * L * CP * 2;  //  8.4 MB each

// ---- helpers ---------------------------------------------------------------
// bf16 A/B fragment (16x32 / 32x16): lane = 16*half + r,
//   elems 0..7  -> K = half*8 + j      (16 contiguous bytes)
//   elems 8..15 -> K = 16 + half*8 + j (16 contiguous bytes)
__device__ __forceinline__ v16bf frag_ld(const __bf16* __restrict__ p) {
  v16bf f;
#pragma unroll
  for (int j = 0; j < 8; ++j) f[j] = p[j];
#pragma unroll
  for (int j = 0; j < 8; ++j) f[8 + j] = p[16 + j];
  return f;
}

__device__ __forceinline__ v8f wmma_bf16(v16bf a, v16bf b, v8f c) {
  return __builtin_amdgcn_wmma_f32_16x16x32_bf16(false, a, false, b,
                                                 (short)0, c, false, false);
}

// 16-byte global->LDS copy: async DMA if the gfx1250 builtin exists.
// Probe result: builtin exists; param0 is AS1 int4* (global), param1 AS3 (LDS).
__device__ __forceinline__ void copy16_g2l(const float* __restrict__ gsrc,
                                           float* __restrict__ ldst) {
#if __has_builtin(__builtin_amdgcn_global_load_async_to_lds_b128)
  __builtin_amdgcn_global_load_async_to_lds_b128(
      (AS1 v4i*)gsrc, (AS3 v4i*)ldst, 0, 0);
#else
  *(float4*)ldst = *(const float4*)gsrc;
#endif
}

__device__ __forceinline__ void wait_async0() {
#if __has_builtin(__builtin_amdgcn_s_wait_asynccnt)
  __builtin_amdgcn_s_wait_asynccnt(0);
#elif __has_builtin(__builtin_amdgcn_global_load_async_to_lds_b128)
  asm volatile("s_wait_asynccnt 0" ::: "memory");
#endif
}

// ---- one-shot f32 -> bf16 conversion ---------------------------------------
__global__ __launch_bounds__(256) void cvt_kernel(const float* __restrict__ s,
                                                  __bf16* __restrict__ d, int n) {
  const int i = (blockIdx.x * 256 + threadIdx.x) * 4;
  if (i + 3 < n) {
    const float4 v = *(const float4*)(s + i);
    d[i + 0] = (__bf16)v.x;
    d[i + 1] = (__bf16)v.y;
    d[i + 2] = (__bf16)v.z;
    d[i + 3] = (__bf16)v.w;
  }
}

// ---- 16x64 per-wave GEMM core: A reused across 4 WMMAs per K-step ----------
__device__ __forceinline__ void gemm16x64(const __bf16* __restrict__ A, int lda,
                                          const __bf16* __restrict__ Bm, int ldb,
                                          int m0, int n0, int K, int lane,
                                          v8f acc[4]) {
  const int mr = lane & 15;
  const int hf = lane >> 4;
  const __bf16* pa  = A  + (size_t)(m0 + mr) * lda + hf * 8;
  const __bf16* pb0 = Bm + (size_t)(n0 +  0 + mr) * ldb + hf * 8;
  const __bf16* pb1 = Bm + (size_t)(n0 + 16 + mr) * ldb + hf * 8;
  const __bf16* pb2 = Bm + (size_t)(n0 + 32 + mr) * ldb + hf * 8;
  const __bf16* pb3 = Bm + (size_t)(n0 + 48 + mr) * ldb + hf * 8;
#pragma unroll 2
  for (int k0 = 0; k0 < K; k0 += 32) {
    const v16bf a  = frag_ld(pa  + k0);
    const v16bf b0 = frag_ld(pb0 + k0);
    const v16bf b1 = frag_ld(pb1 + k0);
    const v16bf b2 = frag_ld(pb2 + k0);
    const v16bf b3 = frag_ld(pb3 + k0);
    acc[0] = wmma_bf16(a, b0, acc[0]);
    acc[1] = wmma_bf16(a, b1, acc[1]);
    acc[2] = wmma_bf16(a, b2, acc[2]);
    acc[3] = wmma_bf16(a, b3, acc[3]);
  }
}

// ---- Kernel 1: xe = x @ W_up^T (M=1024, N=2560, K=1024) --------------------
// 8 waves/block, 16x64 tile/wave -> block covers 16 rows x 512 cols.
__global__ __launch_bounds__(256) void gemm1_kernel(
    const __bf16* __restrict__ x16,   // [M, T]
    const __bf16* __restrict__ wu16,  // [E, T]
    __bf16* __restrict__ xe) {        // [M, E]
  const int lane = threadIdx.x & 31;
  const int wv   = threadIdx.x >> 5;
  const int n0   = blockIdx.x * 512 + wv * 64;
  const int m0   = blockIdx.y * 16;

  v8f acc[4] = {};
  gemm16x64(x16, T, wu16, T, m0, n0, T, lane, acc);

  const int n  = lane & 15;
  const int hf = lane >> 4;
#pragma unroll
  for (int j = 0; j < 4; ++j)
#pragma unroll
    for (int r = 0; r < 8; ++r)
      xe[(size_t)(m0 + hf * 8 + r) * E + n0 + j * 16 + n] = (__bf16)acc[j][r];
}

// ---- Kernel 2: t = xe @ W_proj^T, scatter into q_ws / k_ws -----------------
__global__ __launch_bounds__(256) void gemm2_kernel(
    const __bf16* __restrict__ xe,    // [M, E]
    const __bf16* __restrict__ wp16,  // [F, E]
    __bf16* __restrict__ q_ws,        // [B, H, L, CP]
    __bf16* __restrict__ k_ws) {
  const int lane = threadIdx.x & 31;
  const int wv   = threadIdx.x >> 5;
  const int n0   = blockIdx.x * 512 + wv * 64;
  const int m0   = blockIdx.y * 16;

  v8f acc[4] = {};
  gemm16x64(xe, E, wp16, E, m0, n0, E, lane, acc);

  const int   n      = lane & 15;
  const int   hf     = lane >> 4;
  const float qscale = 0.22360679774997896f;  // 20^-0.5

#pragma unroll
  for (int j = 0; j < 4; ++j) {
#pragma unroll
    for (int r = 0; r < 8; ++r) {
      const int l  = m0 + hf * 8 + r;       // 0..1023
      const int f  = n0 + j * 16 + n;       // 0..5119
      const int bb = l >> 9;
      const int ll = l & (L - 1);
      const int h  = f / (2 * HW);          // / 40
      const int c  = f - h * (2 * HW);      // % 40
      const size_t idx = (((size_t)bb * H + h) * L + ll) * CP;
      const float v = acc[j][r];
      if (c < HW) q_ws[idx + c]        = (__bf16)(v * qscale);
      else        k_ws[idx + (c - HW)] = (__bf16)v;
    }
  }
}

// ---- Kernel 3: scores = q @ k^T + bias, mask, h-innermost store ------------
// Block: 256 threads = 8 waves, one h per wave. Bias tile is DMA'd to LDS
// asynchronously while the WMMA runs. Grid: (L/16, L/16, B*16).
__global__ __launch_bounds__(256) void attn_kernel(
    const __bf16* __restrict__ q_ws,
    const __bf16* __restrict__ k_ws,
    const float*  __restrict__ bias,   // [B, L, L, H]
    const int*    __restrict__ mask,   // [B, L]
    float* __restrict__ out) {         // [B, L, L, H]
  __shared__ float sc[16 * 16 * 8];        // [m][n][h_local]
  __shared__ float bias_lds[16 * 16 * 8];  // same layout

  const int lane = threadIdx.x & 31;
  const int wv   = threadIdx.x >> 5;       // h_local 0..7
  const int k0   = blockIdx.x * 16;
  const int q0   = blockIdx.y * 16;
  const int bb   = blockIdx.z >> 4;
  const int hg   = blockIdx.z & 15;
  const int h    = hg * 8 + wv;

  // Kick off the 8 KB bias tile DMA first: 512 x 16B chunks, 2 per thread.
#pragma unroll
  for (int i = 0; i < 2; ++i) {
    const int chunk = i * 256 + threadIdx.x;   // 0..511
    const int hl4 = chunk & 1;                 // low/high 4 of the 8 h's
    const int nn  = (chunk >> 1) & 15;
    const int mm  = chunk >> 5;
    const float* gsrc =
        bias + ((((size_t)bb * L + q0 + mm) * L) + k0 + nn) * H + hg * 8 + hl4 * 4;
    copy16_g2l(gsrc, &bias_lds[(mm * 16 + nn) * 8 + hl4 * 4]);
  }

  // q.k^T for this wave's h: K=32 covers c=0..31 (c>=20 zero-padded).
  const __bf16* qbase = q_ws + (((size_t)bb * H + h) * L) * CP;
  const __bf16* kbase = k_ws + (((size_t)bb * H + h) * L) * CP;
  const int mr = lane & 15;
  const int hf = lane >> 4;
  const v16bf a = frag_ld(qbase + (size_t)(q0 + mr) * CP + hf * 8);
  const v16bf b = frag_ld(kbase + (size_t)(k0 + mr) * CP + hf * 8);
  v8f acc = {};
  acc = wmma_bf16(a, b, acc);

#pragma unroll
  for (int r = 0; r < 8; ++r)
    sc[((hf * 8 + r) * 16 + mr) * 8 + wv] = acc[r];

  wait_async0();       // this wave's bias DMAs landed in LDS
  __syncthreads();     // everyone's DMAs + scores visible

  // Epilogue: 2048 elems, 8/thread, h innermost -> coalesced 512 MB stream.
#pragma unroll
  for (int i = 0; i < 8; ++i) {
    const int e  = i * 256 + threadIdx.x;
    const int hl = e & 7;
    const int nn = (e >> 3) & 15;
    const int mm = e >> 7;
    const int kk = k0 + nn;
    const size_t oidx =
        ((((size_t)bb * L + q0 + mm) * L) + kk) * H + hg * 8 + hl;
    const float v = sc[(mm * 16 + nn) * 8 + hl] + bias_lds[(mm * 16 + nn) * 8 + hl];
    out[oidx] = mask[bb * L + kk] ? v : 0.0f;
  }
}

// ---------------------------------------------------------------------------
extern "C" void kernel_launch(void* const* d_in, const int* in_sizes, int n_in,
                              void* d_out, int out_size, void* d_ws, size_t ws_size,
                              hipStream_t stream) {
  const float* x     = (const float*)d_in[0];
  const int*   mask  = (const int*)  d_in[1];
  const float* bias  = (const float*)d_in[2];
  const float* Wup   = (const float*)d_in[3];
  const float* Wproj = (const float*)d_in[4];
  float* out = (float*)d_out;

  char* ws = (char*)d_ws;
  __bf16* x16  = (__bf16*)(ws);
  __bf16* wu16 = (__bf16*)(ws + X16_BYTES);
  __bf16* wp16 = (__bf16*)(ws + X16_BYTES + WU16_BYTES);
  __bf16* xe   = (__bf16*)(ws + X16_BYTES + WU16_BYTES + WP16_BYTES);
  __bf16* q_ws = (__bf16*)(ws + X16_BYTES + WU16_BYTES + WP16_BYTES + XE_BYTES);
  __bf16* k_ws = (__bf16*)(ws + X16_BYTES + WU16_BYTES + WP16_BYTES + XE_BYTES + QK_BYTES);
  // ~53 MB of workspace total.

  // Zero q/k pads (c = 20..31 must read 0 in the K=32 WMMA).
  (void)hipMemsetAsync(q_ws, 0, 2 * QK_BYTES, stream);

  // f32 -> bf16 staging (weights/activations are L2-resident afterwards).
  cvt_kernel<<<(M * T / 4 + 255) / 256, 256, 0, stream>>>(x, x16, M * T);
  cvt_kernel<<<(E * T / 4 + 255) / 256, 256, 0, stream>>>(Wup, wu16, E * T);
  cvt_kernel<<<(F * E / 4 + 255) / 256, 256, 0, stream>>>(Wproj, wp16, F * E);

  gemm1_kernel<<<dim3(E / 512, M / 16), 256, 0, stream>>>(x16, wu16, xe);
  gemm2_kernel<<<dim3(F / 512, M / 16), 256, 0, stream>>>(xe, wp16, q_ws, k_ws);
  attn_kernel<<<dim3(L / 16, L / 16, B * 16), 256, 0, stream>>>(q_ws, k_ws, bias, mask, out);
}
